// ODE_RNN_86852828660358
// MI455X (gfx1250) — matrix operand order repaired and checked by reading, not compiled, not run on hardware
//
#include <hip/hip_runtime.h>
#include <hip/hip_bf16.h>
#include <math.h>

// Problem constants (from the reference)
#define BB   256
#define TT   128
#define HH   256
#define DD   256
#define BM   16              // batch rows per workgroup
#define NWG  (BB / BM)       // 16 blocks
#define NT   512             // threads per block
#define NWAVES (NT / 32)     // 16 waves

typedef __attribute__((ext_vector_type(16))) _Float16 v16h;
typedef __attribute__((ext_vector_type(8)))  _Float16 v8h;
typedef __attribute__((ext_vector_type(8)))  float    v8f;

// ---------------------------------------------------------------------------
// Weight fp32 -> fp16 conversion pre-pass (into d_ws).
// Layout in ws (halves): W1h @0, W2h @65536, Wo1h @131072, Whhh @196608.
// ---------------------------------------------------------------------------
__global__ void cvt_weights_kernel(const float* __restrict__ W1,
                                   const float* __restrict__ W2,
                                   const float* __restrict__ Wo1,
                                   const float* __restrict__ Whh,
                                   _Float16* __restrict__ ws) {
    int i = blockIdx.x * blockDim.x + threadIdx.x;
    if (i < 65536) {
        ws[i]           = (_Float16)W1[i];
        ws[65536 + i]   = (_Float16)W2[i];
        ws[131072 + i]  = (_Float16)Wo1[i];
    }
    if (i < 196608) {
        ws[196608 + i]  = (_Float16)Whh[i];
    }
}

// ---------------------------------------------------------------------------
// Fragment helpers: one wave computes one 16x16 tile, K=256 (8 WMMAs).
//   A: LDS fp16 16x256 row-major.  B: global fp16 [n][k] (native for x@W^T).
// ---------------------------------------------------------------------------
__device__ __forceinline__ v8f tile_mma(const _Float16* Xlds,
                                        const _Float16* __restrict__ Wh,
                                        int n0, int lane) {
    const int K    = 256;
    const int col  = lane & 15;
    const int hsel = lane >> 4;
    const int ak   = hsel * 8;
    const int bk   = hsel * 8;
    v8f acc = {};
    const _Float16* wrow = Wh + (size_t)(n0 + col) * K;
    const _Float16* xrow = Xlds + col * K;
#pragma unroll
    for (int kk = 0; kk < K; kk += 32) {
        union { v16h v; v8h h[2]; } a;
        a.h[0] = *(const v8h*)(xrow + kk + ak);
        a.h[1] = *(const v8h*)(xrow + kk + 16 + ak);
        union { v16h v; v8h h[2]; } bu;
        bu.h[0] = *(const v8h*)(wrow + kk + bk);
        bu.h[1] = *(const v8h*)(wrow + kk + 16 + bk);
        v16h b = bu.v;
        acc = __builtin_amdgcn_wmma_f32_16x16x32_f16(
            false, a.v, false, b, (short)0, acc, false, false);
    }
    return acc;
}

// Generic GEMM: Y[16 x N] = X @ Wh^T + bias.
// mode 0 -> dst16 fp16 (no act), mode 2 -> dst32 fp32 (no act).
__device__ __forceinline__ void wg_gemm(const _Float16* Xlds,
                                        const _Float16* __restrict__ Wh,
                                        const float*    __restrict__ bias,
                                        int N,
                                        _Float16* dst16, float* dst32, int dstStride,
                                        int mode, int lane, int wave) {
    const int col  = lane & 15;
    const int mrow = (lane >> 4) * 8;
    const int ntiles = N >> 4;
    for (int jt = wave; jt < ntiles; jt += NWAVES) {
        const int n0 = jt << 4;
        v8f acc = tile_mma(Xlds, Wh, n0, lane);
        const float bv = bias[n0 + col];
        if (mode == 0) {
#pragma unroll
            for (int r = 0; r < 8; ++r)
                dst16[(mrow + r) * dstStride + n0 + col] = (_Float16)(acc[r] + bv);
        } else {
#pragma unroll
            for (int r = 0; r < 8; ++r)
                dst32[(mrow + r) * dstStride + n0 + col] = acc[r] + bv;
        }
    }
}

// RK4-fused GEMM: k = tanh(Mid @ W2^T + b2), fused with the RK4 stage update.
//   stg 0: Acc = k;        Xh = f16(Hf + 0.5*dt*k)
//   stg 1: Acc += 2k;      Xh = f16(Hf + 0.5*dt*k)
//   stg 2: Acc += 2k;      Xh = f16(Hf + dt*k)
//   stg 3: Hf += dt/6*(Acc + k);  Xh = f16(Hf)
__device__ __forceinline__ void wg_gemm_rk4(const _Float16* Mid,
                                            const _Float16* __restrict__ W2h,
                                            const float*    __restrict__ b2g,
                                            float* Hf, float* Acc, _Float16* Xh,
                                            int stg, float dt,
                                            int lane, int wave) {
    const int col  = lane & 15;
    const int mrow = (lane >> 4) * 8;
    for (int jt = wave; jt < (HH >> 4); jt += NWAVES) {
        const int n0 = jt << 4;
        v8f acc = tile_mma(Mid, W2h, n0, lane);
        const float bv = b2g[n0 + col];
#pragma unroll
        for (int r = 0; r < 8; ++r) {
            const int idx = (mrow + r) * HH + n0 + col;
            const float k = tanhf(acc[r] + bv);
            if (stg == 0) {
                Acc[idx] = k;
                Xh[idx]  = (_Float16)(Hf[idx] + 0.5f * dt * k);
            } else if (stg == 1) {
                Acc[idx] += 2.0f * k;
                Xh[idx]   = (_Float16)(Hf[idx] + 0.5f * dt * k);
            } else if (stg == 2) {
                Acc[idx] += 2.0f * k;
                Xh[idx]   = (_Float16)(Hf[idx] + dt * k);
            } else {
                const float hn = Hf[idx] + (dt * (1.0f / 6.0f)) * (Acc[idx] + k);
                Hf[idx] = hn;
                Xh[idx] = (_Float16)hn;
            }
        }
    }
}

__device__ __forceinline__ float sigmoidf_(float x) {
    return 1.0f / (1.0f + __expf(-x));
}

// ---------------------------------------------------------------------------
// Persistent ODE-RNN scan. One block = 16 batch rows, full T loop.
// ---------------------------------------------------------------------------
__global__ __launch_bounds__(NT, 1)
void ode_rnn_kernel(const float* __restrict__ bIn,     // (B,T,2)
                    const float* __restrict__ mIn,     // (B,T)
                    const float* __restrict__ trmIn,   // (B,T)
                    const float* __restrict__ temIn,   // (B,T)
                    const float* __restrict__ h0In,    // (B,H)
                    const float* __restrict__ b1g,     // (D)
                    const float* __restrict__ b2g,     // (H)
                    const float* __restrict__ Wihg,    // (3H,1)
                    const float* __restrict__ bihg,    // (3H)
                    const float* __restrict__ bhhg,    // (3H)
                    const float* __restrict__ bo1g,    // (D)
                    const float* __restrict__ bo2g,    // (1)
                    const float* __restrict__ Wo2g,    // (1,D)
                    const _Float16* __restrict__ W1h,  // (D,H) fp16
                    const _Float16* __restrict__ W2h,  // (H,D) fp16
                    const _Float16* __restrict__ Wo1h, // (D,H) fp16
                    const _Float16* __restrict__ Whhh, // (3H,H) fp16
                    float* __restrict__ outPtr) {      // (B,T)
    extern __shared__ char smem_raw[];
    float*    Hf   = (float*)smem_raw;        // 16x256 hidden state (fp32 master)
    float*    Acc  = Hf   + BM * HH;          // 16x256 RK4 k-accumulator
    float*    Kf   = Acc  + BM * HH;          // 16x256 Wo1 output
    float*    Gh   = Kf   + BM * HH;          // 16x768 GRU gates (gh + b_hh)
    float*    red  = Gh   + BM * 3 * HH;      // 16x16 reduction scratch
    float*    s_bv = red  + 256;              // per-row step scalars
    float*    s_mv  = s_bv  + BM;
    float*    s_trm = s_mv  + BM;
    float*    s_tem = s_trm + BM;
    float*    s_out = s_tem + BM;
    _Float16* Xh  = (_Float16*)(s_out + BM);  // 16x256 fp16 GEMM input (tracks Hf)
    _Float16* Mid = Xh + BM * HH;             // 16x256 fp16 layer-1 output

    const int tid  = threadIdx.x;
    const int lane = tid & 31;
    const int wave = tid >> 5;
    const int rb0  = blockIdx.x * BM;         // first batch row of this block

    // init hidden state (fp32 master + fp16 shadow)
    for (int idx = tid; idx < BM * HH; idx += NT) {
        const float h = h0In[(size_t)rb0 * HH + idx];
        Hf[idx] = h;
        Xh[idx] = (_Float16)h;
    }
    __syncthreads();

    for (int t = 0; t < TT; ++t) {
        // per-step scalars (uniform + per-row)
        const float t1v = bIn[(size_t)t * 2];                       // b[0,t,0]
        const float t0v = (t == 0) ? 0.0f : bIn[(size_t)(t - 1) * 2];
        const float dt  = (t1v - t0v) * 0.5f;                       // / N_SUB
        if (tid < BM) {
            const int b = rb0 + tid;
            s_bv[tid]  = bIn[((size_t)b * TT + t) * 2 + 1];
            s_mv[tid]  = mIn[(size_t)b * TT + t];
            s_trm[tid] = trmIn[(size_t)b * TT + t];
            s_tem[tid] = temIn[(size_t)b * TT + t];
        }
        __syncthreads();

        // ----------------- RK4: 2 substeps x 4 fused stages -----------------
        // Invariant entering each stage: Xh holds f16 of the f() input.
        for (int sub = 0; sub < 2; ++sub) {
            for (int stg = 0; stg < 4; ++stg) {
                wg_gemm(Xh, W1h, b1g, DD, Mid, nullptr, DD, 0, lane, wave);
                __syncthreads();
                wg_gemm_rk4(Mid, W2h, b2g, Hf, Acc, Xh, stg, dt, lane, wave);
                __syncthreads();
            }
        }
        // Hf holds hp (fp32); Xh holds f16(hp).

        // ----------------- output head + GRU gates (both read only Xh) -----
        wg_gemm(Xh, Wo1h, bo1g, DD, nullptr, Kf, DD, 2, lane, wave);
        wg_gemm(Xh, Whhh, bhhg, 3 * HH, nullptr, Gh, 3 * HH, 2, lane, wave);
        __syncthreads();

        // out[b] = tanh(dot(Kf[b,:], Wo2) + bo2); 256 threads participate.
        if (tid < 256) {
            const int r = tid >> 4, c = tid & 15;
            float s = 0.0f;
            for (int d = c; d < DD; d += 16)
                s += Kf[r * DD + d] * Wo2g[d];
            red[r * 16 + c] = s;
        }
        __syncthreads();
        if (tid < 256 && (tid & 15) == 0) {
            const int r = tid >> 4;
            float tot = 0.0f;
#pragma unroll
            for (int i = 0; i < 16; ++i) tot += red[r * 16 + i];
            const float ov = tanhf(tot + bo2g[0]);
            s_out[r] = ov;
            outPtr[(size_t)(rb0 + r) * TT + t] = ov;
        }
        __syncthreads();

        // ----------------- GRU elementwise update of Hf (+ f16 shadow) -----
        for (int idx = tid; idx < BM * HH; idx += NT) {
            const int r = idx >> 8;
            const int h = idx & 255;
            const float hp  = Hf[idx];
            const float ghr = Gh[r * 768 + h];
            const float ghz = Gh[r * 768 + 256 + h];
            const float ghn = Gh[r * 768 + 512 + h];
            const float wir = Wihg[h],       wiz = Wihg[256 + h], win = Wihg[512 + h];
            const float bir = bihg[h],       biz = bihg[256 + h], bin_ = bihg[512 + h];
            const float trm = s_trm[r], tem = s_tem[r], mv = s_mv[r];
            // train branch: x = bv * trm
            const float xtr = s_bv[r] * trm;
            float rr = sigmoidf_(xtr * wir + bir + ghr);
            float zz = sigmoidf_(xtr * wiz + biz + ghz);
            float nn = tanhf(xtr * win + bin_ + rr * ghn);
            const float htr = (1.0f - zz) * nn + zz * hp;
            // test branch: x = out * tem
            const float xte = s_out[r] * tem;
            rr = sigmoidf_(xte * wir + bir + ghr);
            zz = sigmoidf_(xte * wiz + biz + ghz);
            nn = tanhf(xte * win + bin_ + rr * ghn);
            const float hte = (1.0f - zz) * nn + zz * hp;
            const float hnew = trm * htr + tem * hte + (1.0f - mv) * hp;
            Hf[idx] = hnew;
            Xh[idx] = (_Float16)hnew;
        }
        __syncthreads();
    }
}

// ---------------------------------------------------------------------------
extern "C" void kernel_launch(void* const* d_in, const int* in_sizes, int n_in,
                              void* d_out, int out_size, void* d_ws, size_t ws_size,
                              hipStream_t stream) {
    const float* bIn   = (const float*)d_in[0];
    const float* mIn   = (const float*)d_in[1];
    const float* trmIn = (const float*)d_in[2];
    const float* temIn = (const float*)d_in[3];
    const float* h0In  = (const float*)d_in[4];
    const float* W1    = (const float*)d_in[5];
    const float* b1    = (const float*)d_in[6];
    const float* W2    = (const float*)d_in[7];
    const float* b2    = (const float*)d_in[8];
    const float* Wih   = (const float*)d_in[9];
    const float* Whh   = (const float*)d_in[10];
    const float* bih   = (const float*)d_in[11];
    const float* bhh   = (const float*)d_in[12];
    const float* Wo1   = (const float*)d_in[13];
    const float* bo1   = (const float*)d_in[14];
    const float* Wo2   = (const float*)d_in[15];
    const float* bo2   = (const float*)d_in[16];
    float* out = (float*)d_out;

    _Float16* ws16 = (_Float16*)d_ws;
    const _Float16* W1h  = ws16;
    const _Float16* W2h  = ws16 + 65536;
    const _Float16* Wo1h = ws16 + 131072;
    const _Float16* Whhh = ws16 + 196608;

    // 1) convert weights fp32 -> fp16 into workspace
    cvt_weights_kernel<<<(196608 + 255) / 256, 256, 0, stream>>>(W1, W2, Wo1, Whh, ws16);

    // 2) persistent batch-parallel scan; 16 blocks x 512 threads (16 waves)
    const size_t smem = (size_t)(BM * HH * 3 + BM * 3 * HH + 256 + 5 * BM) * sizeof(float)
                      + (size_t)(2 * BM * HH) * sizeof(_Float16);
    ode_rnn_kernel<<<NWG, NT, smem, stream>>>(
        bIn, mIn, trmIn, temIn, h0In, b1, b2, Wih, bih, bhh, bo1, bo2, Wo2,
        W1h, W2h, Wo1h, Whhh, out);
}
